// Gray_Conv2d_32126355374719
// MI455X (gfx1250) — compile-verified, matrix operand order checked by reading, not run on hardware
//
#include <hip/hip_runtime.h>

typedef __attribute__((ext_vector_type(16))) _Float16 v16h;
typedef __attribute__((ext_vector_type(8)))  float    v8f;
typedef __attribute__((ext_vector_type(4)))  float    v4f;
typedef __attribute__((ext_vector_type(4)))  unsigned int u32x4;
typedef __attribute__((ext_vector_type(8)))  int      i32x8;
typedef __attribute__((ext_vector_type(4)))  int      i32x4;

#define C_IN   150
#define C_OUT  128
#define KSZ    5
#define HH     56
#define WW     56
#define DFULL  (C_IN * KSZ * KSZ)     // 3750
#define NCHUNK 125                    // 25 kernel positions * 5 channel-blocks of 32
#define CHUNK_BYTES 8192              // 8 subtiles * 32 lanes * 32B
#define SQRT34 0.86602540378443864676f

// ---------------------------------------------------------------------------
// Pack sqrt(w_c)*weight[o, c*25+p] into per-lane B-fragment order (f16).
// Chunk q = p*5 + cb covers channels [cb*32, cb*32+32) (zero-padded >=150) at
// kernel position p.  Lane (ncol=lane&15, half=lane>>4) of subtile s holds
// column o = s*16+ncol with K elements {half*8+0..7, 16+half*8+0..7}.
// Layout: pk[((q*8+s)*32 + lane)*16 + e]  (32B per lane).
// ---------------------------------------------------------------------------
__global__ __launch_bounds__(256) void pack_weight_kernel(
    const float* __restrict__ w, _Float16* __restrict__ pk) {
  const int q    = blockIdx.x;          // 0..124
  const int p    = q / 5;
  const int cb   = q % 5;
  const int s    = threadIdx.x >> 5;    // 0..7
  const int lane = threadIdx.x & 31;
  const int ncol = lane & 15;
  const int half = lane >> 4;
  const int o    = s * 16 + ncol;

  v16h vals;
#pragma unroll
  for (int e = 0; e < 16; ++e) {
    const int c = cb * 32 + half * 8 + ((e < 8) ? e : (16 + (e - 8)));
    float v = 0.0f;
    if (c < C_IN) {
      const float sc = (c < 75) ? 0.5f : SQRT34;
      v = sc * w[(size_t)o * DFULL + c * 25 + p];
    }
    vals[e] = (_Float16)v;
  }
  *(v16h*)(pk + (((size_t)(q * 8 + s) * 32 + lane) << 4)) = vals;
}

// y2[o] = sum_d w_d * weight[o,d]^2  (exact f32)
__global__ __launch_bounds__(128) void y2_kernel(
    const float* __restrict__ w, float* __restrict__ y2) {
  const int o = threadIdx.x;
  float acc = 0.0f;
  for (int c = 0; c < C_IN; ++c) {
    const float wc = (c < 75) ? 0.25f : 0.75f;
    float s = 0.0f;
#pragma unroll
    for (int p = 0; p < 25; ++p) {
      const float t = w[(size_t)o * DFULL + c * 25 + p];
      s += t * t;
    }
    acc += wc * s;
  }
  y2[o] = acc;
}

// ---------------------------------------------------------------------------
// TDM: 1-D descriptor moving one 8KB B-chunk (2048 dwords) Global -> LDS.
// D# group0: count=1 | lds_addr | global_addr(57b) | type=2.
// D# group1: data_size=4B, tensor_dim0=tile_dim0=2048, dim0_stride=2048.
// Groups 2/3 zero (<=2D tensor).
// ---------------------------------------------------------------------------
__device__ __forceinline__ void tdm_load_chunk(const _Float16* pk, int q,
                                               unsigned lds_off) {
  const unsigned long long ga =
      (unsigned long long)(uintptr_t)pk + (unsigned long long)q * CHUNK_BYTES;
  u32x4 g0;
  g0[0] = 1u;                                   // count=1, user descriptor
  g0[1] = lds_off;                              // lds_addr (bytes)
  g0[2] = (unsigned)ga;                         // global_addr[31:0]
  g0[3] = (unsigned)((ga >> 32) & 0x1FFFFFFull) // global_addr[56:32]
        | (2u << 30);                           // type = 2 ("image")
  i32x8 g1;
  g1[0] = 0x00020000;                           // data_size = 2 (4 bytes)
  g1[1] = (int)(0x0800u << 16);                 // tensor_dim0 = 2048 (lo16)
  g1[2] = (int)(1u << 16);                      // tensor_dim1 = 1
  g1[3] = (int)(0x0800u << 16);                 // tile_dim0 = 2048
  g1[4] = 0;                                    // tile_dim1/2 unused
  g1[5] = 2048;                                 // tensor_dim0_stride
  g1[6] = 0;
  g1[7] = 0;
  const i32x4 z4 = {0, 0, 0, 0};
#if defined(__clang_major__) && (__clang_major__ >= 23)
  const i32x8 z8 = {0, 0, 0, 0, 0, 0, 0, 0};
  __builtin_amdgcn_tensor_load_to_lds(g0, g1, z4, z4, z8, 0);
#else
  __builtin_amdgcn_tensor_load_to_lds(g0, g1, z4, z4, 0);
#endif
}

// ---------------------------------------------------------------------------
// Main implicit-GEMM distance kernel.
// Block = (n, oh), 4 waves; wave handles 16 consecutive ow x all 128 channels.
// B chunks double-buffered in LDS via TDM; fragments read with ds_load_b128.
// ---------------------------------------------------------------------------
__global__ __launch_bounds__(128) void dist_wmma_kernel(
    const float* __restrict__ x, const _Float16* __restrict__ pk,
    const float* __restrict__ y2, float* __restrict__ out) {
  __shared__ __attribute__((aligned(32))) _Float16 lds_b[2][CHUNK_BYTES / 2];

  const int n    = blockIdx.x / HH;
  const int oh   = blockIdx.x % HH;
  const int wave = threadIdx.x >> 5;
  const int lane = threadIdx.x & 31;
  const int m    = lane & 15;           // A row this lane feeds
  const int half = lane >> 4;
  const int ow   = wave * 16 + m;       // 0..63 (>=56 masked at store)

  const unsigned lds0 = (unsigned)(uintptr_t)&lds_b[0][0];
  const unsigned lds1 = (unsigned)(uintptr_t)&lds_b[1][0];

  v8f acc[8];
#pragma unroll
  for (int s = 0; s < 8; ++s)
#pragma unroll
    for (int r = 0; r < 8; ++r) acc[s][r] = 0.0f;

  float px2 = 0.0f;                      // lane-partial of sum w_c * x^2
  const float* __restrict__ xn = x + (size_t)n * C_IN * HH * WW;

  if (wave == 0) tdm_load_chunk(pk, 0, lds0);   // prologue: chunk 0 -> buf 0

  for (int p = 0; p < 25; ++p) {
    const int ki = p / 5, kj = p % 5;
    const int ih = oh + ki - 2;
    const int iw = ow + kj - 2;
    const bool inb = ((unsigned)ih < (unsigned)HH) && ((unsigned)iw < (unsigned)WW);
    const int ihc = (ih < 0) ? 0 : ((ih > HH - 1) ? HH - 1 : ih);
    const int iwc = (iw < 0) ? 0 : ((iw > WW - 1) ? WW - 1 : iw);
    const float* __restrict__ xb = xn + (size_t)ihc * WW + iwc;   // c = 0 base

#pragma unroll
    for (int cb = 0; cb < 5; ++cb) {
      const int q  = p * 5 + cb;
      const int c0 = cb * 32 + half * 8;

      // --- sync: buf[q&1] ready; everyone done with buf[(q+1)&1] ---
      if (wave == 0) __builtin_amdgcn_s_wait_tensorcnt(0);
      __syncthreads();
      if (wave == 0 && (q + 1) < NCHUNK)        // prefetch next chunk
        tdm_load_chunk(pk, q + 1, ((q + 1) & 1) ? lds1 : lds0);

      // --- A gather: 16 channels at fixed (ih,iw), clamped + cndmask ---
      v16h afrag;
#pragma unroll
      for (int e = 0; e < 16; ++e) {
        const int c  = c0 + ((e < 8) ? e : (16 + (e - 8)));
        const int cc = (c < C_IN) ? c : (C_IN - 1);               // clamp pad
        float v = xb[(size_t)cc * (HH * WW)];                     // always load
        v = (inb && (c < C_IN)) ? v : 0.0f;                       // cndmask
        v *= (c < 75) ? 0.5f : SQRT34;                            // sqrt(w_c)
        px2 += v * v;
        afrag[e] = (_Float16)v;
      }

      // --- B fragments from LDS (2x ds_load_b128 each), then WMMA chain ---
      const v16h* __restrict__ bbuf = (const v16h*)&lds_b[q & 1][0];
      v16h bfrag[8];
#pragma unroll
      for (int s = 0; s < 8; ++s) bfrag[s] = bbuf[s * 32 + lane];
#pragma unroll
      for (int s = 0; s < 8; ++s)
        acc[s] = __builtin_amdgcn_wmma_f32_16x16x32_f16(
            false, afrag, false, bfrag[s], (short)0, acc[s], false, false);
    }
  }

  // x2 per row: lanes m and m+16 hold disjoint K halves -> combine, broadcast.
  const float px2sum = px2 + __shfl_xor(px2, 16, 32);
  float xrow[8];
#pragma unroll
  for (int r = 0; r < 8; ++r) xrow[r] = __shfl(px2sum, half * 8 + r, 32);

  // C/D layout: lane column o = s*16 + (lane&15); VGPR r -> row M = r + 8*half.
  const int ocol   = lane & 15;
  const int owbase = wave * 16 + half * 8;   // 8 consecutive ow per lane
  const bool wvalid = (owbase < WW);

#pragma unroll
  for (int s = 0; s < 8; ++s) {
    const int o = s * 16 + ocol;
    const float y2o = y2[o];
    v4f d0, d1;
#pragma unroll
    for (int r = 0; r < 8; ++r) {
      const float d2 = xrow[r] + y2o - 2.0f * acc[s][r];
      const float dv = __builtin_sqrtf(fmaxf(d2, 0.0f));
      if (r < 4) d0[r] = dv; else d1[r - 4] = dv;
    }
    if (wvalid) {
      const size_t oidx =
          ((((size_t)n * C_OUT) + o) * HH + oh) * WW + owbase;    // 16B aligned
      *(v4f*)(out + oidx)     = d0;
      *(v4f*)(out + oidx + 4) = d1;
    }
  }
}

extern "C" void kernel_launch(void* const* d_in, const int* in_sizes, int n_in,
                              void* d_out, int out_size, void* d_ws, size_t ws_size,
                              hipStream_t stream) {
  const float* x = (const float*)d_in[0];   // [8,150,56,56] f32
  const float* w = (const float*)d_in[1];   // [128,3750]    f32
  float* out = (float*)d_out;               // [8,128,56,56] f32

  _Float16* pk = (_Float16*)d_ws;                               // 1,024,000 B
  float* y2 = (float*)((char*)d_ws + (size_t)NCHUNK * CHUNK_BYTES);

  pack_weight_kernel<<<NCHUNK, 256, 0, stream>>>(w, pk);
  y2_kernel<<<1, 128, 0, stream>>>(w, y2);
  dist_wmma_kernel<<<8 * HH, 128, 0, stream>>>(x, pk, y2, out);
}
